// ComplexLSTM_9268539425402
// MI455X (gfx1250) — matrix operand order repaired and checked by reading, not compiled, not run on hardware
//
#include <hip/hip_runtime.h>

typedef __attribute__((ext_vector_type(16))) __bf16 v16bf;
typedef __attribute__((ext_vector_type(8)))  __bf16 v8bf;
typedef __attribute__((ext_vector_type(8)))  float  v8f;
typedef unsigned int u32;

#define Bc 32      // batch
#define Ic 128     // input features
#define Hc 256     // hidden
#define Tc 2048    // timesteps
#define KT 384     // Ic + Hc (weight row length in LDS)
#define ROWS 256   // gate rows per workgroup slice (4 gates * 64 hidden)

static __device__ __forceinline__ float fexp(float x) {
    return __builtin_amdgcn_exp2f(x * 1.4426950408889634f);
}
static __device__ __forceinline__ float sigmoid_(float x) {
    return __builtin_amdgcn_rcpf(1.0f + fexp(-x));
}
static __device__ __forceinline__ float tanh_(float x) {
    float e = fexp(-2.0f * x);
    return (1.0f - e) * __builtin_amdgcn_rcpf(1.0f + e);
}

static __device__ __forceinline__ v16bf load_frag(const __bf16* p0, const __bf16* p1) {
    v8bf a = *reinterpret_cast<const v8bf*>(p0);
    v8bf b = *reinterpret_cast<const v8bf*>(p1);
    return __builtin_shufflevector(a, b, 0,1,2,3,4,5,6,7,8,9,10,11,12,13,14,15);
}

#define WMMA_BF16(A, Bm, C) \
    __builtin_amdgcn_wmma_f32_16x16x32_bf16(false, (A), false, (Bm), (short)0, (C), false, false)

// ---------------------------------------------------------------------------
// Stage 0a: zero the complex output (atomics accumulate into it).
__global__ void zero_out_kernel(float* __restrict__ out, long n) {
    long stride = (long)gridDim.x * blockDim.x;
    for (long i = (long)blockIdx.x * blockDim.x + threadIdx.x; i < n; i += stride)
        out[i] = 0.0f;
}

// ---------------------------------------------------------------------------
// Stage 0b: convert weights f32->bf16, build bias sums, zero h buffers, reset counters.
__global__ void stage0_kernel(const float* __restrict__ wihr_f, const float* __restrict__ whhr_f,
                              const float* __restrict__ wihi_f, const float* __restrict__ whhi_f,
                              const float* __restrict__ bihr, const float* __restrict__ bhhr,
                              const float* __restrict__ bihi, const float* __restrict__ bhhi,
                              __bf16* __restrict__ wihr, __bf16* __restrict__ whhr,
                              __bf16* __restrict__ wihi, __bf16* __restrict__ whhi,
                              float* __restrict__ bsumr, float* __restrict__ bsumi,
                              __bf16* __restrict__ hglb, u32* __restrict__ counters) {
    const long NW1 = 4 * Hc * Ic;      // 131072
    const long NW2 = 4 * Hc * Hc;      // 262144
    const long NH  = 4L * 2 * Bc * Hc; // 65536
    const long total = 2 * (NW1 + NW2) + 2 * 1024 + NH + 64;
    long stride = (long)gridDim.x * blockDim.x;
    for (long i = (long)blockIdx.x * blockDim.x + threadIdx.x; i < total; i += stride) {
        long j = i;
        if (j < NW1) { wihr[j] = (__bf16)wihr_f[j]; continue; } j -= NW1;
        if (j < NW2) { whhr[j] = (__bf16)whhr_f[j]; continue; } j -= NW2;
        if (j < NW1) { wihi[j] = (__bf16)wihi_f[j]; continue; } j -= NW1;
        if (j < NW2) { whhi[j] = (__bf16)whhi_f[j]; continue; } j -= NW2;
        if (j < 1024) { bsumr[j] = bihr[j] + bhhr[j]; continue; } j -= 1024;
        if (j < 1024) { bsumi[j] = bihi[j] + bhhi[j]; continue; } j -= 1024;
        if (j < NH)   { hglb[j] = (__bf16)0.0f; continue; } j -= NH;
        if (j < 4)    { counters[j] = 0u; }
    }
}

// ---------------------------------------------------------------------------
// Main persistent kernel: 16 workgroups = 4 LSTMs x 4 hidden slices.
// Critical path per step = 8 h-dependent K-chunks (32 WMMA) + activations +
// h publish + barrier. x staging, x-part GEMM (16 WMMA) and output atomics
// are executed in the barrier shadow for the NEXT step.
__global__ __launch_bounds__(256, 1)
void lstm_main(const float* __restrict__ xr, const float* __restrict__ xi,
               const __bf16* __restrict__ wihr, const __bf16* __restrict__ whhr,
               const __bf16* __restrict__ wihi, const __bf16* __restrict__ whhi,
               const float* __restrict__ bsumr, const float* __restrict__ bsumi,
               __bf16* __restrict__ hglb, u32* __restrict__ counters,
               float* __restrict__ out) {
    const int tid   = threadIdx.x;
    const int wg    = blockIdx.x;
    const int lstm  = wg >> 2;    // 0: xr*Wr(+re) 1: xi*Wi(-re) 2: xi*Wr(+im) 3: xr*Wi(+im)
    const int slice = wg & 3;     // 64 hidden units per slice

    const float*  x    = (lstm == 0 || lstm == 3) ? xr : xi;
    const __bf16* wih  = (lstm == 0 || lstm == 2) ? wihr : wihi;
    const __bf16* whh  = (lstm == 0 || lstm == 2) ? whhr : whhi;
    const float*  bsum = (lstm == 0 || lstm == 2) ? bsumr : bsumi;
    const float   sgn  = (lstm == 1) ? -1.0f : 1.0f;
    const int     ioff = (lstm >= 2) ? 1 : 0;

    __shared__ __align__(16) __bf16 wbuf[ROWS * KT];    // 192 KB weight slice
    __shared__ __align__(16) __bf16 xbuf[2][Bc * Ic];   // 2 x 8 KB x_t staging
    __shared__ __align__(16) __bf16 hbuf[Bc * Hc];      // 16 KB h_prev staging

    // ---- one-time weight slice preload: one gate-row per thread ----
    {
        const int r = tid;                 // 0..255
        const int gate = r >> 6, hh = r & 63;
        const int g = gate * Hc + slice * 64 + hh;
        const u32* sI = (const u32*)(wih + (size_t)g * Ic);
        const u32* sH = (const u32*)(whh + (size_t)g * Hc);
        u32* d = (u32*)(wbuf + r * KT);
        #pragma unroll 8
        for (int k = 0; k < Ic / 2; ++k) d[k] = sI[k];
        #pragma unroll 8
        for (int k = 0; k < Hc / 2; ++k) d[Ic / 2 + k] = sH[k];
    }
    // ---- zero hbuf (h_0 = 0), stage x_0 ----
    {
        u32* hz = (u32*)hbuf;
        #pragma unroll
        for (int k = 0; k < 16; ++k) hz[tid + k * 256] = 0u;
        #pragma unroll
        for (int k = 0; k < 16; ++k) {
            const int idx = tid + k * 256;
            const int b = idx >> 7, i2 = idx & 127;
            xbuf[0][b * Ic + i2] = (__bf16)x[((size_t)(b * Ic + i2)) * Tc + 0];
        }
    }

    const int wave = tid >> 5, lane = tid & 31;
    const int laneM = lane & 15, hiHalf = lane >> 4;
    const int m_tile = wave & 1, hn = wave >> 1;   // 2 batch tiles x 4 hidden tiles
    const int hid = slice * 64 + hn * 16 + laneM;  // this lane's hidden column

    float bias[4];
    #pragma unroll
    for (int g = 0; g < 4; ++g) bias[g] = bsum[g * Hc + hid];

    const int arowoff = (m_tile * 16 + laneM);
    const __bf16* arowh = hbuf + arowoff * Hc;
    const __bf16* brow0 = wbuf + (0 * 64 + hn * 16 + laneM) * KT;
    const __bf16* brow1 = wbuf + (1 * 64 + hn * 16 + laneM) * KT;
    const __bf16* brow2 = wbuf + (2 * 64 + hn * 16 + laneM) * KT;
    const __bf16* brow3 = wbuf + (3 * 64 + hn * 16 + laneM) * KT;

    v8f cst = {};                      // cell state stays in VGPRs (D layout)
    v8f a0 = {}, a1 = {}, a2 = {}, a3 = {};
    __syncthreads();

    // ---- prologue: x-part GEMM for t=0 ----
    {
        const __bf16* arowx = xbuf[0] + arowoff * Ic;
        #pragma unroll
        for (int kc = 0; kc < 4; ++kc) {
            const int aoff0 = kc * 32 + hiHalf * 8,  aoff1 = aoff0 + 16;
            const int boff0 = kc * 32 + hiHalf * 8, boff1 = boff0 + 16;
            v16bf af = load_frag(arowx + aoff0, arowx + aoff1);
            a0 = WMMA_BF16(af, load_frag(brow0 + boff0, brow0 + boff1), a0);
            a1 = WMMA_BF16(af, load_frag(brow1 + boff0, brow1 + boff1), a1);
            a2 = WMMA_BF16(af, load_frag(brow2 + boff0, brow2 + boff1), a2);
            a3 = WMMA_BF16(af, load_frag(brow3 + boff0, brow3 + boff1), a3);
        }
    }

    for (int t = 0; t < Tc; ++t) {
        // ---- critical path: h-dependent K-chunks (Whh portion of wbuf rows) ----
        #pragma unroll
        for (int kc = 0; kc < 8; ++kc) {
            const int aoff0 = kc * 32 + hiHalf * 8,       aoff1 = aoff0 + 16;
            const int boff0 = Ic + kc * 32 + hiHalf * 8, boff1 = boff0 + 16;
            v16bf af = load_frag(arowh + aoff0, arowh + aoff1);
            a0 = WMMA_BF16(af, load_frag(brow0 + boff0, brow0 + boff1), a0);
            a1 = WMMA_BF16(af, load_frag(brow1 + boff0, brow1 + boff1), a1);
            a2 = WMMA_BF16(af, load_frag(brow2 + boff0, brow2 + boff1), a2);
            a3 = WMMA_BF16(af, load_frag(brow3 + boff0, brow3 + boff1), a3);
        }

        // ---- activations + cell update + publish h slice ----
        float hv[8];
        __bf16* hdst = hglb + ((size_t)(lstm * 2 + ((t + 1) & 1))) * Bc * Hc;
        #pragma unroll
        for (int e = 0; e < 8; ++e) {
            const float iv = sigmoid_(a0[e] + bias[0]);
            const float fv = sigmoid_(a1[e] + bias[1]);
            const float gv = tanh_   (a2[e] + bias[2]);
            const float ov = sigmoid_(a3[e] + bias[3]);
            const float c  = fv * cst[e] + iv * gv;
            cst[e] = c;
            hv[e] = ov * tanh_(c);
            const int b = m_tile * 16 + hiHalf * 8 + e;
            hdst[b * Hc + hid] = (__bf16)hv[e];
        }
        __threadfence();       // h slice visible device-wide
        __syncthreads();
        if (tid == 0)          // signal EARLY so sibling WGs can proceed
            __hip_atomic_fetch_add(counters + lstm, 1u,
                                   __ATOMIC_RELEASE, __HIP_MEMORY_SCOPE_AGENT);

        // ================= barrier shadow (independent of h_{t+1}) ==========
        // output atomics for step t
        #pragma unroll
        for (int e = 0; e < 8; ++e) {
            const int b = m_tile * 16 + hiHalf * 8 + e;
            const size_t oidx = (((size_t)(b * Hc + hid)) * Tc + t) * 2 + ioff;
            __hip_atomic_fetch_add(out + oidx, sgn * hv[e],
                                   __ATOMIC_RELAXED, __HIP_MEMORY_SCOPE_AGENT);
        }
        // stage x_{t+1} (scattered stride-T loads, latency hidden here)
        const int tn = (t + 1 < Tc) ? t + 1 : t;
        const int par = (t + 1) & 1;
        #pragma unroll
        for (int k = 0; k < 16; ++k) {
            const int idx = tid + k * 256;
            const int b = idx >> 7, i2 = idx & 127;
            const float* xp = x + ((size_t)(b * Ic + i2)) * Tc + tn;
            xbuf[par][b * Ic + i2] = (__bf16)(*xp);
            if ((t & 31) == 0 && t + 32 < Tc) __builtin_prefetch(xp + 32, 0, 0);
        }
        __syncthreads();       // xbuf[par] ready
        // x-part GEMM for step t+1 (Wih portion of wbuf rows)
        a0 = (v8f){}; a1 = (v8f){}; a2 = (v8f){}; a3 = (v8f){};
        {
            const __bf16* arowx = xbuf[par] + arowoff * Ic;
            #pragma unroll
            for (int kc = 0; kc < 4; ++kc) {
                const int aoff0 = kc * 32 + hiHalf * 8,  aoff1 = aoff0 + 16;
                const int boff0 = kc * 32 + hiHalf * 8, boff1 = boff0 + 16;
                v16bf af = load_frag(arowx + aoff0, arowx + aoff1);
                a0 = WMMA_BF16(af, load_frag(brow0 + boff0, brow0 + boff1), a0);
                a1 = WMMA_BF16(af, load_frag(brow1 + boff0, brow1 + boff1), a1);
                a2 = WMMA_BF16(af, load_frag(brow2 + boff0, brow2 + boff1), a2);
                a3 = WMMA_BF16(af, load_frag(brow3 + boff0, brow3 + boff1), a3);
            }
        }
        // ====================================================================

        // ---- wait for all 4 sibling WGs of this LSTM ----
        if (tid == 0) {
            const u32 tgt = 4u * (u32)(t + 1);
            while (__hip_atomic_load(counters + lstm,
                                     __ATOMIC_ACQUIRE, __HIP_MEMORY_SCOPE_AGENT) < tgt)
                __builtin_amdgcn_s_sleep(1);
        }
        __syncthreads();

        // ---- stage h_{t+1} into LDS via async global->LDS DMA path ----
        {
            const char* hsrcB = (const char*)(hglb +
                                ((size_t)(lstm * 2 + ((t + 1) & 1))) * Bc * Hc);
            const u32 ldsbase = (u32)(uintptr_t)(&hbuf[0]);
            #pragma unroll
            for (int k = 0; k < 4; ++k) {
                const int idx = tid + k * 256;          // 1024 x 16B granules
                const u32 ldsoff = ldsbase + (u32)idx * 16u;
                const u32 voff   = (u32)idx * 16u;
                asm volatile("global_load_async_to_lds_b128 %0, %1, %2"
                             :: "v"(ldsoff), "v"(voff), "s"(hsrcB) : "memory");
            }
            asm volatile("s_wait_asynccnt 0x0" ::: "memory");
        }
        __syncthreads();
    }
}

// ---------------------------------------------------------------------------
extern "C" void kernel_launch(void* const* d_in, const int* in_sizes, int n_in,
                              void* d_out, int out_size, void* d_ws, size_t ws_size,
                              hipStream_t stream) {
    (void)in_sizes; (void)n_in; (void)ws_size;
    const float* xr     = (const float*)d_in[0];
    const float* xi     = (const float*)d_in[1];
    const float* wihr_f = (const float*)d_in[2];
    const float* whhr_f = (const float*)d_in[3];
    const float* bihr   = (const float*)d_in[4];
    const float* bhhr   = (const float*)d_in[5];
    const float* wihi_f = (const float*)d_in[6];
    const float* whhi_f = (const float*)d_in[7];
    const float* bihi   = (const float*)d_in[8];
    const float* bhhi   = (const float*)d_in[9];
    float* out = (float*)d_out;

    char* ws = (char*)d_ws;
    __bf16* wihr = (__bf16*)(ws);                       // 256 KB
    __bf16* whhr = (__bf16*)(ws + 262144);              // 512 KB
    __bf16* wihi = (__bf16*)(ws + 786432);
    __bf16* whhi = (__bf16*)(ws + 1048576);
    float*  bsumr = (float*)(ws + 1572864);
    float*  bsumi = (float*)(ws + 1576960);
    __bf16* hglb  = (__bf16*)(ws + 1581056);            // [4][2][B][H] bf16
    u32*    counters = (u32*)(ws + 1712128);

    zero_out_kernel<<<8192, 256, 0, stream>>>(out, (long)out_size);
    stage0_kernel<<<2048, 256, 0, stream>>>(wihr_f, whhr_f, wihi_f, whhi_f,
                                            bihr, bhhr, bihi, bhhi,
                                            wihr, whhr, wihi, whhi,
                                            bsumr, bsumi, hglb, counters);
    lstm_main<<<16, 256, 0, stream>>>(xr, xi, wihr, whhr, wihi, whhi,
                                      bsumr, bsumi, hglb, counters, out);
}
